// MultiHeadedAttention_39178691674803
// MI455X (gfx1250) — matrix operand order repaired and checked by reading, not compiled, not run on hardware
//
#include <hip/hip_runtime.h>

// ---------------- constants ----------------
constexpr int NB  = 8;     // batch
constexpr int NS  = 2048;  // sequence
constexpr int ND  = 20;    // model dim
constexpr int NH  = 4;     // heads
constexpr int NDK = 5;     // head dim
constexpr int DKP = 8;     // padded head dim for Q/K (dim5 = mask/log2e channel)
constexpr int DVP = 8;     // padded head dim rows for transposed V (row5 = ones)
constexpr int WAVES = 4;   // waves per attention block (wave32)

typedef __bf16 v16bf __attribute__((ext_vector_type(16)));
typedef float  v8f   __attribute__((ext_vector_type(8)));

__device__ __forceinline__ unsigned short f2bf(float f) {
    const __bf16 h = (__bf16)f;                       // native RNE conversion
    return __builtin_bit_cast(unsigned short, h);
}

__device__ __forceinline__ float fast_exp2(float x) {
    return __builtin_amdgcn_exp2f(x);                 // v_exp_f32
}

// =====================================================================
// Kernel 1: 9 projections -> bf16.
//   Qp[b][a][h][s][0..4] = proj0 * (1/sqrt(5))*log2(e),  [5] = log2(e)
//   Kp[b][x][h][s][0..4] = proj1,                        [5] = mask ? 0 : -1e9
//   Vt[b][x][h][d][s]    = proj2 (d<5),  row d=5 = 1.0 (row-sum channel)
// =====================================================================
__global__ __launch_bounds__(256) void proj_kernel(
    const float* __restrict__ q, const float* __restrict__ k, const float* __restrict__ v,
    const int* __restrict__ mask,
    const float* __restrict__ W0, const float* __restrict__ b0,
    const float* __restrict__ W1, const float* __restrict__ b1,
    const float* __restrict__ W2, const float* __restrict__ b2,
    unsigned short* __restrict__ Qp, unsigned short* __restrict__ Kp,
    unsigned short* __restrict__ Vt)
{
    __shared__ float sW[3][ND * ND];
    __shared__ float sBv[3][ND];
    const float* Ws[3] = {W0, W1, W2};
    const float* Bs[3] = {b0, b1, b2};
    for (int p = 0; p < 3; ++p) {
        for (int i = threadIdx.x; i < ND * ND; i += blockDim.x) sW[p][i] = Ws[p][i];
        if (threadIdx.x < ND) sBv[p][threadIdx.x] = Bs[p][threadIdx.x];
    }
    __syncthreads();

    const int row = blockIdx.x * blockDim.x + threadIdx.x;
    if (row >= NB * 3 * NS) return;
    const int b   = row / (3 * NS);
    const int rem = row - b * 3 * NS;
    const int src = rem / NS;
    const int s   = rem - src * NS;

    const float* xs = (src == 0 ? q : (src == 1 ? k : v)) + (size_t)(b * NS + s) * ND;
    float xr[ND];
#pragma unroll
    for (int d = 0; d < ND; ++d) xr[d] = xs[d];

    const float L2E  = 1.4426950408889634f;          // log2(e)
    const float rs5  = 0.44721359549995794f * L2E;   // 1/sqrt(DK) * log2(e)
    const int   base = (b * 3 + src) * NH;
    const unsigned short qExtra = f2bf(L2E);
    const unsigned short kExtra = (mask[b * NS + s] != 0) ? (unsigned short)0
                                                          : f2bf(-1.0e9f);

    // Q (p=0) and K (p=1): row-major [.., s, DKP] bf16
#pragma unroll
    for (int p = 0; p < 2; ++p) {
        unsigned short* dst = (p == 0 ? Qp : Kp);
        const float sc = (p == 0 ? rs5 : 1.0f);
        const unsigned short extra = (p == 0 ? qExtra : kExtra);
        for (int h = 0; h < NH; ++h) {
            unsigned int u[4] = {0u, 0u, 0u, 0u};
            for (int dk = 0; dk < NDK; ++dk) {
                const int o = h * NDK + dk;
                float acc = sBv[p][o];
#pragma unroll
                for (int d = 0; d < ND; ++d) acc += xr[d] * sW[p][o * ND + d];
                acc *= sc;
                u[dk >> 1] |= (unsigned int)f2bf(acc) << ((dk & 1) * 16);
            }
            u[2] |= (unsigned int)extra << 16;   // dim 5 channel
            *(uint4*)(dst + ((size_t)(base + h) * NS + s) * DKP) =
                make_uint4(u[0], u[1], u[2], u[3]);
        }
    }

    // V transposed: [.., h, dk, s]; dk=5 -> 1.0 (row-sum channel), dk=6,7 -> 0
    for (int h = 0; h < NH; ++h) {
        for (int dk = 0; dk < DVP; ++dk) {
            unsigned short val;
            if (dk < NDK) {
                const int o = h * NDK + dk;
                float acc = sBv[2][o];
#pragma unroll
                for (int d = 0; d < ND; ++d) acc += xr[d] * sW[2][o * ND + d];
                val = f2bf(acc);
            } else {
                val = (dk == NDK) ? (unsigned short)0x3F80 : (unsigned short)0; // 1.0 / 0
            }
            Vt[((size_t)(base + h) * DVP + dk) * NS + s] = val;
        }
    }
}

// =====================================================================
// Kernel 2: flash attention, transposed dataflow.
//   scores^T = K_tile x Q  (mask folded into dim-5 channel)
//   O^T      = V^T x P^T   (row-sum folded into V row 5)
//   softmax state is per-lane scalar; no LDS; no exec-mask dance in loop.
// =====================================================================
__global__ __launch_bounds__(WAVES * 32) void attn_kernel(
    const unsigned short* __restrict__ Qp, const unsigned short* __restrict__ Kp,
    const unsigned short* __restrict__ Vt, float* __restrict__ outA)
{
    const int lane  = threadIdx.x & 31;
    const int wave  = threadIdx.x >> 5;
    const int h     = blockIdx.y;
    const int b     = blockIdx.z;
    const int qbase = (blockIdx.x * WAVES + wave) * 16;

    const int half = lane >> 4;   // 0: lanes 0-15, 1: lanes 16-31
    const int ln   = lane & 15;   // this lane's query column q = qbase + ln

    // lane masks (VGPR-resident, replaces exec-guarded loads)
    const unsigned int lmK = (half == 0) ? ~0u : 0u;   // K/Q frags live in lower half
    const unsigned int lmV = (ln < DVP)  ? ~0u : 0u;   // V^T has 8 real dim rows

    float Oacc[NDK];
#pragma unroll
    for (int r = 0; r < NDK; ++r) Oacc[r] = 0.0f;

    for (int a = 0; a < 3; ++a) {
        // ---- Q B-frag: lane column q = qbase+ln, dims 0-15 (8 real + zeros) ----
        v16bf qf;
        {
            const uint4 qr = *(const uint4*)(
                Qp + (((size_t)(b * 3 + a) * NH + h) * NS + qbase + ln) * DKP);
            unsigned int* qu = (unsigned int*)&qf;
            qu[0] = qr.x & lmK; qu[1] = qr.y & lmK;
            qu[2] = qr.z & lmK; qu[3] = qr.w & lmK;
            qu[4] = 0u; qu[5] = 0u; qu[6] = 0u; qu[7] = 0u;
        }

        for (int xi = 0; xi < 2; ++xi) {
            const int x = (a + 1 + xi) % 3;
            const unsigned short* Kb = Kp + ((size_t)(b * 3 + x) * NH + h) * NS * DKP;
            const unsigned short* Vb = Vt + ((size_t)(b * 3 + x) * NH + h) * DVP * NS
                                          + (size_t)(ln & 7) * NS;  // clamped dim row

            float m = -INFINITY;   // running max (log2 domain), per query column
            v8f acc;               // O^T accumulator: reg r = dim r, col = q; reg5 = sum(p)
#pragma unroll
            for (int r = 0; r < 8; ++r) acc[r] = 0.0f;
            v8f zc;
#pragma unroll
            for (int r = 0; r < 8; ++r) zc[r] = 0.0f;

#pragma unroll 2
            for (int k0 = 0; k0 < NS; k0 += 32) {
                // ---- K A-frags (rows = keys): two 16-key tiles, maskless loads ----
                v16bf kf0, kf1;
                {
                    const uint4 k0r = *(const uint4*)(Kb + (size_t)(k0 + ln) * DKP);
                    const uint4 k1r = *(const uint4*)(Kb + (size_t)(k0 + 16 + ln) * DKP);
                    unsigned int* u0 = (unsigned int*)&kf0;
                    unsigned int* u1 = (unsigned int*)&kf1;
                    u0[0] = k0r.x & lmK; u0[1] = k0r.y & lmK;
                    u0[2] = k0r.z & lmK; u0[3] = k0r.w & lmK;
                    u0[4] = 0u; u0[5] = 0u; u0[6] = 0u; u0[7] = 0u;
                    u1[0] = k1r.x & lmK; u1[1] = k1r.y & lmK;
                    u1[2] = k1r.z & lmK; u1[3] = k1r.w & lmK;
                    u1[4] = 0u; u1[5] = 0u; u1[6] = 0u; u1[7] = 0u;
                }
                // scores^T: D[key][q]; mask bias added by dim-5 channel
                v8f s0 = __builtin_amdgcn_wmma_f32_16x16x32_bf16(
                    false, kf0, false, qf, (short)0, zc, false, false);
                v8f s1 = __builtin_amdgcn_wmma_f32_16x16x32_bf16(
                    false, kf1, false, qf, (short)0, zc, false, false);

                // ---- per-lane max over 16 local keys, then partner half ----
                float mm = fmaxf(s0[0], s1[0]);
#pragma unroll
                for (int r = 1; r < 8; ++r) mm = fmaxf(mm, fmaxf(s0[r], s1[r]));
                mm = fmaxf(mm, __shfl_xor(mm, 16, 32));
                const float mnew = fmaxf(m, mm);
                const float corr = fast_exp2(m - mnew);
                m = mnew;

                // ---- p = exp2(s - m), packed bf16 pairs (keys in order) ----
                unsigned int pk0[4], pk1[4];
#pragma unroll
                for (int i = 0; i < 4; ++i) {
                    const float a0 = fast_exp2(s0[2 * i]     - mnew);
                    const float a1 = fast_exp2(s0[2 * i + 1] - mnew);
                    const float c0 = fast_exp2(s1[2 * i]     - mnew);
                    const float c1 = fast_exp2(s1[2 * i + 1] - mnew);
                    pk0[i] = (unsigned int)f2bf(a0) | ((unsigned int)f2bf(a1) << 16);
                    pk1[i] = (unsigned int)f2bf(c0) | ((unsigned int)f2bf(c1) << 16);
                }

                // ---- rescale accumulator by per-lane scalar ----
#pragma unroll
                for (int r = 0; r < 8; ++r) acc[r] *= corr;

                // ---- build P^T B-frag: half-wave swap, 4 shuffles ----
                v16bf pb;
                unsigned int* pbu = (unsigned int*)&pb;
#pragma unroll
                for (int i = 0; i < 4; ++i) {
                    const unsigned int t = (half == 0) ? pk1[i] : pk0[i];
                    const unsigned int e = __shfl_xor(t, 16, 32);
                    pbu[i]     = (half == 0) ? pk0[i] : e;   // keys 0-7  / 16-23
                    pbu[4 + i] = (half == 0) ? e : pk1[i];   // keys 8-15 / 24-31
                }

                // ---- V^T A-frag: row = dim (ln&7), 16 contiguous keys per half ----
                v16bf vf;
                {
                    const unsigned short* vp = Vb + k0 + 16 * half;
                    const uint4 v0r = *(const uint4*)vp;          // K {0-7} / {16-23}
                    const uint4 v1r = *(const uint4*)(vp + 8);    // K {8-15} / {24-31}
                    unsigned int* vu = (unsigned int*)&vf;
                    vu[0] = v0r.x & lmV; vu[1] = v0r.y & lmV;
                    vu[2] = v0r.z & lmV; vu[3] = v0r.w & lmV;
                    const uint4 v2r = *(const uint4*)(vp + 8);    // K {8-15} / {24-31}
                    vu[4] = v2r.x & lmV; vu[5] = v2r.y & lmV;
                    vu[6] = v2r.z & lmV; vu[7] = v2r.w & lmV;
                    (void)v1r;
                }
                acc = __builtin_amdgcn_wmma_f32_16x16x32_bf16(
                    false, vf, false, pb, (short)0, acc, false, false);
            } // key blocks

            // ---- finalize: row-sum sits in acc[5] (valid for lanes 0-15) ----
            const float inv = 1.0f / acc[5];
#pragma unroll
            for (int r = 0; r < NDK; ++r) Oacc[r] += acc[r] * inv;
        } // xi
    } // a

    // store: lane ln holds query column qbase+ln, dims 0-4 in regs 0-4
    if (half == 0) {
#pragma unroll
        for (int r = 0; r < NDK; ++r) {
            outA[((size_t)(b * NS) + qbase + ln) * ND + h * NDK + r] = Oacc[r];
        }
    }
}

// =====================================================================
// Kernel 3: final 20x20 projection out = outA @ W3^T + b3
// =====================================================================
__global__ __launch_bounds__(256) void oproj_kernel(
    const float* __restrict__ oat, const float* __restrict__ W3,
    const float* __restrict__ b3, float* __restrict__ out)
{
    __shared__ float sW[ND * ND];
    __shared__ float sB[ND];
    for (int i = threadIdx.x; i < ND * ND; i += blockDim.x) sW[i] = W3[i];
    if (threadIdx.x < ND) sB[threadIdx.x] = b3[threadIdx.x];
    __syncthreads();

    const int rs = blockIdx.x * blockDim.x + threadIdx.x;
    if (rs >= NB * NS) return;
    float xr[ND];
#pragma unroll
    for (int i = 0; i < ND; ++i) xr[i] = oat[(size_t)rs * ND + i];
#pragma unroll
    for (int o = 0; o < ND; ++o) {
        float acc = sB[o];
#pragma unroll
        for (int i = 0; i < ND; ++i) acc += xr[i] * sW[o * ND + i];
        out[(size_t)rs * ND + o] = acc;
    }
}

// =====================================================================
extern "C" void kernel_launch(void* const* d_in, const int* in_sizes, int n_in,
                              void* d_out, int out_size, void* d_ws, size_t ws_size,
                              hipStream_t stream)
{
    (void)in_sizes; (void)n_in; (void)out_size; (void)ws_size;
    const float* q   = (const float*)d_in[0];
    const float* k   = (const float*)d_in[1];
    const float* v   = (const float*)d_in[2];
    const int*   msk = (const int*)  d_in[3];
    const float* W0  = (const float*)d_in[4];
    const float* b0  = (const float*)d_in[5];
    const float* W1  = (const float*)d_in[6];
    const float* b1  = (const float*)d_in[7];
    const float* W2  = (const float*)d_in[8];
    const float* b2  = (const float*)d_in[9];
    const float* W3  = (const float*)d_in[10];
    const float* b3  = (const float*)d_in[11];

    // workspace carve (offsets are 256B-aligned)
    constexpr size_t QP_BYTES = (size_t)NB * 3 * NH * NS * DKP * 2; // 3,145,728
    constexpr size_t VT_BYTES = (size_t)NB * 3 * NH * DVP * NS * 2; // 3,145,728
    char* ws = (char*)d_ws;
    unsigned short* Qp   = (unsigned short*)(ws);
    unsigned short* Kp   = (unsigned short*)(ws + QP_BYTES);
    unsigned short* Vt   = (unsigned short*)(ws + 2 * QP_BYTES);
    float*          outA = (float*)(ws + 2 * QP_BYTES + VT_BYTES);

    // 1) projections (+ mask/log2e/ones channels)
    {
        const int rows = NB * 3 * NS;
        proj_kernel<<<(rows + 255) / 256, 256, 0, stream>>>(
            q, k, v, msk, W0, b0, W1, b1, W2, b2, Qp, Kp, Vt);
    }
    // 2) flash attention over all 6 (a,x) pairs
    {
        dim3 grid(NS / (16 * WAVES), NH, NB);
        attn_kernel<<<grid, WAVES * 32, 0, stream>>>(Qp, Kp, Vt, outA);
    }
    // 3) output projection
    {
        const int rows = NB * NS;
        oproj_kernel<<<(rows + 255) / 256, 256, 0, stream>>>(outA, W3, b3, (float*)d_out);
    }
}
